// UpTransformer_15298673508397
// MI455X (gfx1250) — matrix operand order repaired and checked by reading, not compiled, not run on hardware
//
#include <hip/hip_runtime.h>
#include <stdint.h>

// =============================================================================
// UpTransformer fused pipeline for MI455X (gfx1250, wave32, WMMA bf16)
//
//   1) prep_kernel     : fold BN into scale/bias, swizzle all weights into
//                        WMMA A-fragment layout (bf16) in workspace
//   2) tr_bf16_kernel  : query/key_feat (B,128,N) f32 -> (B,N,128) bf16
//   3) knn_kernel      : top-20 nearest neighbors (LDS-tiled, register sort)
//   4) qkv_kernel      : q/k/v = W(64x128) @ X via v_wmma_f32_16x16x32_bf16
//   5) attn_kernel     : fragment pool staged to LDS via
//                        global_load_async_to_lds_b128 (ASYNCcnt), then fused
//                        gather + pos-MLP + attn-MLP + convT + online softmax
//                        over K + aggregation + final conv + residual
// =============================================================================

constexpr int cB = 4, cN = 4096, cN2 = 4096;
constexpr int cCIN = 128, cDIM = 64, cKNN = 20, cUP = 2;
constexpr int cHPOS = 64, cHATT = 256;
constexpr float cEPS = 1e-5f;

// ---------------- workspace layout (bytes) ----------------
constexpr size_t OFF_IDX  = 0;                                        // int[B*N*KNN]
constexpr size_t OFF_QT   = OFF_IDX + (size_t)cB*cN*cKNN*4;           // f32[B*N*64]
constexpr size_t OFF_KT   = OFF_QT  + (size_t)cB*cN*64*4;             // f32[B*N2*64]
constexpr size_t OFF_VT   = OFF_KT  + (size_t)cB*cN2*64*4;            // f32[B*N2*64]
constexpr size_t OFF_XQT  = OFF_VT  + (size_t)cB*cN2*64*4;            // bf16[B*N*128]
constexpr size_t OFF_XKT  = OFF_XQT + (size_t)cB*cN*128*2;            // bf16[B*N2*128]
constexpr size_t OFF_WQF  = OFF_XKT + (size_t)cB*cN2*128*2;           // 16 frags
constexpr size_t OFF_WKF  = OFF_WQF  + 16*1024;
constexpr size_t OFF_WVF  = OFF_WKF  + 16*1024;
// the next 4 regions are contiguous: 120 frags = 120KB staged to LDS by attn
constexpr size_t OFF_PW2F = OFF_WVF  + 16*1024;                       // 8 frags
constexpr size_t OFF_AW1F = OFF_PW2F + 8*1024;                        // 32 frags
constexpr size_t OFF_AWTF = OFF_AW1F + 32*1024;                       // 64 frags
constexpr size_t OFF_WEF  = OFF_AWTF + 64*1024;                       // 16 frags
constexpr size_t OFF_PS   = OFF_WEF  + 16*1024;                       // f32[64]
constexpr size_t OFF_PB   = OFF_PS + 256;                             // f32[64]
constexpr size_t OFF_AS   = OFF_PB + 256;                             // f32[256]
constexpr size_t OFF_AB   = OFF_AS + 1024;                            // f32[256]
// total ~21.4 MiB

constexpr int FRAG_POOL = 120 * 512;      // 120 frags x 512 bf16 = 120 KB

// ---------------- CDNA5 WMMA types ----------------
typedef __attribute__((ext_vector_type(16))) __bf16 bf16x16;
typedef __attribute__((ext_vector_type(8)))  float  f32x8;

__device__ __forceinline__ f32x8 wmma_bf16(bf16x16 a, bf16x16 b, f32x8 c) {
  // D(16x16 f32) = A(16x32 bf16) * B(32x16 bf16) + C
  return __builtin_amdgcn_wmma_f32_16x16x32_bf16(false, a, false, b,
                                                 (short)0, c, false, false);
}

__device__ __forceinline__ __bf16 f2bf(float x) {      // RNE f32->bf16
  union { float f; unsigned u; } a; a.f = x;
  unsigned r = a.u + 0x7FFFu + ((a.u >> 16) & 1u);
  union { unsigned short s; __bf16 b; } o; o.s = (unsigned short)(r >> 16);
  return o.b;
}

// load a 32-byte B/A fragment (generic: works for LDS or global addresses)
__device__ __forceinline__ bf16x16 ld_frag32B(const __bf16* p) {
  union { uint4 u[2]; bf16x16 v; } fu;
  const uint4* q = reinterpret_cast<const uint4*>(p);
  fu.u[0] = q[0]; fu.u[1] = q[1];
  return fu.v;
}
// A-fragment: 32 lanes x 16 bf16 = 1KB
__device__ __forceinline__ bf16x16 ld_afrag(const __bf16* base, int fragIdx) {
  int lane = threadIdx.x & 31;
  return ld_frag32B(base + (size_t)fragIdx * 512 + lane * 16);
}

// =============================================================================
// 1) prep: fold BN, build WMMA A-fragments (16-bit A 16x32, contiguous k = 16*(lane/16) + j)
// =============================================================================
__device__ void fill_frags(const float* W, int lda, __bf16* dst,
                           int mtiles, int ksteps, int t, int mode) {
  int total = mtiles * ksteps * 512;
  for (int e = t; e < total; e += 256) {
    int f = e >> 9, r = e & 511, lane = r >> 4, j = r & 15;
    int mt = f / ksteps, ks = f - mt * ksteps;
    int v = j >> 1;
    int k = ks * 32 + ((lane >= 16) ? 16 : 0)
          + (v << 1) + (j & 1);
    int row = mt * 16 + (lane & 15);
    float val;
    if (mode == 0) val = W[(size_t)row * lda + k];
    else           val = W[((size_t)k * cDIM + (row >> 1)) * cUP + (row & 1)]; // awt^T
    dst[e] = f2bf(val);
  }
}

__global__ __launch_bounds__(256)
void prep_kernel(const float* wq, const float* wk, const float* wv,
                 const float* pw2, const float* aw1, const float* awt,
                 const float* we,
                 const float* pb1, const float* pg, const float* pbt,
                 const float* pm, const float* pv,
                 const float* ab1, const float* ag, const float* abt2,
                 const float* am, const float* av, char* ws) {
  int t = threadIdx.x;
  float* pS = (float*)(ws + OFF_PS); float* pB = (float*)(ws + OFF_PB);
  float* aS = (float*)(ws + OFF_AS); float* aB = (float*)(ws + OFF_AB);
  if (t < cHPOS) {
    float inv = pg[t] / sqrtf(pv[t] + cEPS);
    pS[t] = inv; pB[t] = (pb1[t] - pm[t]) * inv + pbt[t];
  }
  if (t < cHATT) {
    float inv = ag[t] / sqrtf(av[t] + cEPS);
    aS[t] = inv; aB[t] = (ab1[t] - am[t]) * inv + abt2[t];
  }
  fill_frags(wq,  128, (__bf16*)(ws + OFF_WQF),  4, 4, t, 0);
  fill_frags(wk,  128, (__bf16*)(ws + OFF_WKF),  4, 4, t, 0);
  fill_frags(wv,  128, (__bf16*)(ws + OFF_WVF),  4, 4, t, 0);
  fill_frags(pw2,  64, (__bf16*)(ws + OFF_PW2F), 4, 2, t, 0);
  fill_frags(aw1,  64, (__bf16*)(ws + OFF_AW1F),16, 2, t, 0);
  fill_frags(awt,   0, (__bf16*)(ws + OFF_AWTF), 8, 8, t, 1);
  fill_frags(we,   64, (__bf16*)(ws + OFF_WEF),  8, 2, t, 0);
}

// =============================================================================
// 2) transpose + convert: (B,128,N) f32 -> (B,N,128) bf16
// =============================================================================
__global__ __launch_bounds__(256)
void tr_bf16_kernel(const float* __restrict__ src, __bf16* __restrict__ dst) {
  int i = blockIdx.x * 256 + threadIdx.x;          // over B*N*16
  if (i >= cB * cN * 16) return;
  int bn = i >> 4, c0 = (i & 15) * 8;
  int b = bn / cN, nn = bn - b * cN;
  union { __bf16 h[8]; uint4 u; } pk;
#pragma unroll
  for (int c = 0; c < 8; ++c)
    pk.h[c] = f2bf(src[((size_t)b * cCIN + c0 + c) * cN + nn]);
  *reinterpret_cast<uint4*>(dst + (size_t)bn * cCIN + c0) = pk.u;
}

// =============================================================================
// 3) KNN: thread-per-query, LDS-tiled support points, register top-20
// =============================================================================
__global__ __launch_bounds__(256)
void knn_kernel(const float* __restrict__ pos1, const float* __restrict__ pos2,
                int* __restrict__ idxW) {
  __shared__ float spx[512], spy[512], spz[512], ss2[512];
  int b = blockIdx.y;
  int qn = blockIdx.x * 256 + threadIdx.x;
  float px = pos1[(size_t)(b * 3 + 0) * cN + qn];
  float py = pos1[(size_t)(b * 3 + 1) * cN + qn];
  float pz = pos1[(size_t)(b * 3 + 2) * cN + qn];
  float s1 = px * px + py * py + pz * pz;
  float bd[cKNN]; int bi[cKNN];
#pragma unroll
  for (int j = 0; j < cKNN; ++j) { bd[j] = 3.0e38f; bi[j] = 0; }
  for (int t0 = 0; t0 < cN2; t0 += 512) {
    for (int i = threadIdx.x; i < 512; i += 256) {
      float x = pos2[(size_t)(b * 3 + 0) * cN2 + t0 + i];
      float y = pos2[(size_t)(b * 3 + 1) * cN2 + t0 + i];
      float z = pos2[(size_t)(b * 3 + 2) * cN2 + t0 + i];
      spx[i] = x; spy[i] = y; spz[i] = z; ss2[i] = x * x + y * y + z * z;
    }
    __syncthreads();
    for (int j = 0; j < 512; ++j) {
      float d = s1 + ss2[j] - 2.0f * (px * spx[j] + py * spy[j] + pz * spz[j]);
      if (d < bd[cKNN - 1]) {                 // stable ascending insert
        float cd = d; int ci = t0 + j;
#pragma unroll
        for (int u = 0; u < cKNN; ++u) {
          if (cd < bd[u]) {
            float td = bd[u]; int ti = bi[u];
            bd[u] = cd; bi[u] = ci; cd = td; ci = ti;
          }
        }
      }
    }
    __syncthreads();
  }
  int* op = idxW + ((size_t)b * cN + qn) * cKNN;
#pragma unroll
  for (int j = 0; j < cKNN; ++j) op[j] = bi[j];
}

// =============================================================================
// 4) q/k/v projection: OUT(B,N,64) = W(64x128) @ X + bias,  WMMA bf16
// =============================================================================
__global__ __launch_bounds__(256)
void qkv_kernel(const __bf16* __restrict__ XT, const __bf16* __restrict__ WF,
                const float* __restrict__ bias, float* __restrict__ OUT) {
  int b = blockIdx.y;
  int lane = threadIdx.x & 31, w = threadIdx.x >> 5;
  int hi = (lane >= 16) ? 1 : 0;
  int n = blockIdx.x * 128 + w * 16 + (lane & 15);
  bf16x16 bfr[4];
#pragma unroll
  for (int ks = 0; ks < 4; ++ks)   // B frag: 16 consecutive k for column n
    bfr[ks] = ld_frag32B(XT + ((size_t)b * cN + n) * cCIN + ks * 32 + hi * 16);
#pragma unroll
  for (int mt = 0; mt < 4; ++mt) {
    f32x8 acc = {};
#pragma unroll
    for (int ks = 0; ks < 4; ++ks) {
      __builtin_prefetch(WF + (size_t)(mt * 4 + ks + 1) * 512, 0, 3);
      acc = wmma_bf16(ld_afrag(WF, mt * 4 + ks), bfr[ks], acc);
    }
    int c0 = mt * 16 + hi * 8;
    union { float f[4]; float4 v; } o0, o1;
#pragma unroll
    for (int v = 0; v < 4; ++v) { o0.f[v] = acc[v] + bias[c0 + v];
                                  o1.f[v] = acc[v + 4] + bias[c0 + v + 4]; }
    float* op = OUT + ((size_t)b * cN + n) * 64 + c0;
    *reinterpret_cast<float4*>(op)     = o0.v;
    *reinterpret_cast<float4*>(op + 4) = o1.v;
  }
}

// =============================================================================
// 5) fused attention kernel. grid (N/16, B), 256 threads (8 waves).
//    Fragment pool (120KB) staged global->LDS with async-to-LDS ops; per
//    k-step the 16-wide point-tile = {16 queries at fixed k}, so each lane
//    accumulates online-softmax stats for a fixed (channel,query) in regs.
// =============================================================================
__global__ __launch_bounds__(256)
void attn_kernel(const float* __restrict__ pos1, const float* __restrict__ pos2,
                 const float* __restrict__ query,
                 const float* __restrict__ pw1, const float* __restrict__ pb2,
                 const float* __restrict__ abt, const float* __restrict__ be,
                 const char* __restrict__ ws, float* __restrict__ out) {
  const int*    idxW = (const int*)  (ws + OFF_IDX);
  const float*  qT   = (const float*)(ws + OFF_QT);
  const float*  kT   = (const float*)(ws + OFF_KT);
  const float*  vT   = (const float*)(ws + OFF_VT);
  const float*  pSg  = (const float*)(ws + OFF_PS);
  const float*  pBg  = (const float*)(ws + OFF_PB);
  const float*  aSg  = (const float*)(ws + OFF_AS);
  const float*  aBg  = (const float*)(ws + OFF_AB);

  // ---- LDS ----
  __shared__ __bf16 s_frags[FRAG_POOL];    // 120KB: pw2|aw1|awt|we fragments
  __shared__ float  s_pw1[cHPOS][4];
  __shared__ float  s_pS[cHPOS], s_pB[cHPOS], s_pb2[cHPOS];
  __shared__ float  s_aS[cHATT], s_aB[cHATT];
  __shared__ float  s_abt[cDIM], s_be[cCIN];
  __shared__ float  s_qkrel[16][68];       // q - gather(k), [query][chan]
  __shared__ float  s_pe[16][68];          // pos embedding, [query][chan]
  __shared__ float  s_vg[16][68];          // gather(v)+pe,  [query][chan]
  __shared__ __bf16 s_h[16][72];           // bf16 h, B-frag layout [n][k]
  __shared__ __bf16 s_a[16][264];          // bf16 activations, [n][k]
  __shared__ __bf16 s_agg[32][72];         // softmax-aggregated values [col][k]

  int tid = threadIdx.x;

  // ---- stage 120KB fragment pool into LDS via async-to-LDS (ASYNCcnt) ----
  {
    const char* src = ws + OFF_PW2F;             // uniform 64-bit base -> SGPRs
    unsigned dst0 = (unsigned)(uintptr_t)(&s_frags[0]);
#pragma unroll 1
    for (int i = tid; i < FRAG_POOL / 8; i += 256) {   // 16B per op
      unsigned lo = dst0 + (unsigned)i * 16u;
      unsigned vo = (unsigned)i * 16u;
      asm volatile("global_load_async_to_lds_b128 %0, %1, %2"
                   :: "v"(lo), "v"(vo), "s"(src) : "memory");
    }
    asm volatile("s_wait_asynccnt 0" ::: "memory");
  }
  const __bf16* pw2L = s_frags;                  // 8 frags
  const __bf16* aw1L = s_frags + 8  * 512;       // 32 frags
  const __bf16* awtL = s_frags + 40 * 512;       // 64 frags
  const __bf16* weL  = s_frags + 104 * 512;      // 16 frags

  if (tid < cHPOS) {
    s_pw1[tid][0] = pw1[tid * 3 + 0];
    s_pw1[tid][1] = pw1[tid * 3 + 1];
    s_pw1[tid][2] = pw1[tid * 3 + 2];
    s_pS[tid] = pSg[tid]; s_pB[tid] = pBg[tid];
    s_pb2[tid] = pb2[tid]; s_abt[tid] = abt[tid];
  }
  if (tid < cCIN)  s_be[tid] = be[tid];
  if (tid < cHATT) { s_aS[tid] = aSg[tid]; s_aB[tid] = aBg[tid]; }
  __syncthreads();

  int b = blockIdx.y;
  int q0 = blockIdx.x * 16;
  int lane = tid & 31, w = tid >> 5;
  int hi = (lane >= 16) ? 1 : 0;
  int qL = lane & 15;
  int qg = q0 + qL;

  float rmax[8], rsum[8], rwv[8];
#pragma unroll
  for (int v = 0; v < 8; ++v) { rmax[v] = -3.0e38f; rsum[v] = 0.f; rwv[v] = 0.f; }

  for (int kk = 0; kk < cKNN; ++kk) {
    // ---- phase A/B (concurrent across waves) ----
    if (w < 4) {
      // pos-MLP layer 1 (3->64, VALU) directly into B-fragment registers,
      // then G2 = pw2(64x64) @ pe1 via WMMA; wave w owns m-tile w.
      int id = idxW[((size_t)b * cN + qg) * cKNN + kk];
      float pr0 = pos1[(size_t)(b*3+0)*cN + qg] - pos2[(size_t)(b*3+0)*cN2 + id];
      float pr1 = pos1[(size_t)(b*3+1)*cN + qg] - pos2[(size_t)(b*3+1)*cN2 + id];
      float pr2 = pos1[(size_t)(b*3+2)*cN + qg] - pos2[(size_t)(b*3+2)*cN2 + id];
      bf16x16 p1f[2];
#pragma unroll
      for (int ks = 0; ks < 2; ++ks) {
        bf16x16 fv;
#pragma unroll
        for (int j = 0; j < 16; ++j) {
          int ch = ks * 32 + hi * 16 + j;
          float d = s_pw1[ch][0]*pr0 + s_pw1[ch][1]*pr1 + s_pw1[ch][2]*pr2;
          d = fmaxf(d * s_pS[ch] + s_pB[ch], 0.f);
          fv[j] = f2bf(d);
        }
        p1f[ks] = fv;
      }
      f32x8 acc = {};
      acc = wmma_bf16(ld_afrag(pw2L, w * 2 + 0), p1f[0], acc);
      acc = wmma_bf16(ld_afrag(pw2L, w * 2 + 1), p1f[1], acc);
      int c0 = w * 16 + hi * 8;
#pragma unroll
      for (int v = 0; v < 8; ++v) s_pe[qL][c0 + v] = acc[v] + s_pb2[c0 + v];
    } else {
      // gather q - k and raw v for this (query, kk)
      int t = tid - 128;
      int qq = t & 15, c0 = (t >> 4) * 8;
      int qgg = q0 + qq;
      int id = idxW[((size_t)b * cN + qgg) * cKNN + kk];
      const float* qp = qT + ((size_t)b * cN  + qgg) * 64 + c0;
      const float* kp = kT + ((size_t)b * cN2 + id ) * 64 + c0;
      const float* vp = vT + ((size_t)b * cN2 + id ) * 64 + c0;
#pragma unroll
      for (int i = 0; i < 8; ++i) {
        s_qkrel[qq][c0 + i] = qp[i] - kp[i];
        s_vg[qq][c0 + i]    = vp[i];
      }
    }
    __syncthreads();

    // ---- phase C: h = qk_rel + pe (bf16), vg += pe ----
    {
      int i0 = tid * 4;
      int qq = i0 >> 6, c = i0 & 63;
#pragma unroll
      for (int i = 0; i < 4; ++i) {
        float pe = s_pe[qq][c + i];
        s_h[qq][c + i] = f2bf(s_qkrel[qq][c + i] + pe);
        s_vg[qq][c + i] += pe;
      }
    }
    __syncthreads();

    // ---- phase D: a = relu(bn(aw1(256x64) @ h)) ; wave w -> m-tiles 2w,2w+1
    bf16x16 hb0 = ld_frag32B(&s_h[qL][hi * 16]);
    bf16x16 hb1 = ld_frag32B(&s_h[qL][32 + hi * 16]);
#pragma unroll
    for (int m = 0; m < 2; ++m) {
      int mt = 2 * w + m;
      f32x8 acc = {};
      acc = wmma_bf16(ld_afrag(aw1L, mt * 2 + 0), hb0, acc);
      acc = wmma_bf16(ld_afrag(aw1L, mt * 2 + 1), hb1, acc);
#pragma unroll
      for (int v = 0; v < 8; ++v) {
        int ch = mt * 16 + hi * 8 + v;
        s_a[qL][ch] = f2bf(fmaxf(acc[v] * s_aS[ch] + s_aB[ch], 0.f));
      }
    }
    __syncthreads();

    // ---- phase E: logits = awt^T(128x256) @ a ; online softmax over kk ----
    {
      f32x8 acc = {};
#pragma unroll
      for (int ks = 0; ks < 8; ++ks) {
        acc = wmma_bf16(ld_afrag(awtL, w * 8 + ks),
                        ld_frag32B(&s_a[qL][ks * 32 + hi * 16]), acc);
      }
#pragma unroll
      for (int v = 0; v < 8; ++v) {
        int m = w * 16 + hi * 8 + v;
        int o = m >> 1;
        float x  = acc[v] + s_abt[o];
        float mo = rmax[v];
        float mn = fmaxf(mo, x);
        float corr = __expf(mo - mn);
        float e    = __expf(x - mn);
        rsum[v] = rsum[v] * corr + e;
        rwv[v]  = rwv[v]  * corr + e * s_vg[qL][o];
        rmax[v] = mn;
      }
    }
    __syncthreads();   // protect s_* before next iteration overwrites
  }

  // ---- phase F: normalized aggregation -> bf16 B-operand layout ----
#pragma unroll
  for (int v = 0; v < 8; ++v) {
    int m = w * 16 + hi * 8 + v;
    int o = m >> 1, r = m & 1;
    s_agg[qL * 2 + r][o] = f2bf(rwv[v] / rsum[v]);
  }
  __syncthreads();

  // ---- phase G: y = we(128x64) @ agg + be + identity ----
#pragma unroll
  for (int nt = 0; nt < 2; ++nt) {
    int col = nt * 16 + qL;
    bf16x16 b0 = ld_frag32B(&s_agg[col][hi * 16]);
    bf16x16 b1 = ld_frag32B(&s_agg[col][32 + hi * 16]);
    f32x8 acc = {};
    acc = wmma_bf16(ld_afrag(weL, w * 2 + 0), b0, acc);
    acc = wmma_bf16(ld_afrag(weL, w * 2 + 1), b1, acc);
#pragma unroll
    for (int v = 0; v < 8; ++v) {
      int cin = w * 16 + hi * 8 + v;
      float y = acc[v] + s_be[cin]
              + query[((size_t)b * cCIN + cin) * cN + q0 + (col >> 1)];
      out[((size_t)b * cCIN + cin) * (cN * cUP) + q0 * 2 + col] = y;
    }
  }
}

// =============================================================================
// launcher
// =============================================================================
extern "C" void kernel_launch(void* const* d_in, const int* in_sizes, int n_in,
                              void* d_out, int out_size, void* d_ws, size_t ws_size,
                              hipStream_t stream) {
  const float* pos1 = (const float*)d_in[0];
  const float* query= (const float*)d_in[1];
  const float* pos2 = (const float*)d_in[2];
  const float* key  = (const float*)d_in[3];
  const float* wq   = (const float*)d_in[4];  const float* bq  = (const float*)d_in[5];
  const float* wk   = (const float*)d_in[6];  const float* bk  = (const float*)d_in[7];
  const float* wv   = (const float*)d_in[8];  const float* bv  = (const float*)d_in[9];
  const float* pw1  = (const float*)d_in[10]; const float* pb1 = (const float*)d_in[11];
  const float* pg   = (const float*)d_in[12]; const float* pbt = (const float*)d_in[13];
  const float* pm   = (const float*)d_in[14]; const float* pvv = (const float*)d_in[15];
  const float* pw2  = (const float*)d_in[16]; const float* pb2 = (const float*)d_in[17];
  const float* aw1  = (const float*)d_in[18]; const float* ab1 = (const float*)d_in[19];
  const float* ag   = (const float*)d_in[20]; const float* abt2= (const float*)d_in[21];
  const float* am   = (const float*)d_in[22]; const float* av  = (const float*)d_in[23];
  const float* awt  = (const float*)d_in[24]; const float* abt = (const float*)d_in[25];
  const float* we   = (const float*)d_in[26]; const float* be  = (const float*)d_in[27];

  char* ws = (char*)d_ws;

  prep_kernel<<<1, 256, 0, stream>>>(wq, wk, wv, pw2, aw1, awt, we,
                                     pb1, pg, pbt, pm, pvv,
                                     ab1, ag, abt2, am, av, ws);
  tr_bf16_kernel<<<(cB * cN * 16) / 256, 256, 0, stream>>>(query, (__bf16*)(ws + OFF_XQT));
  tr_bf16_kernel<<<(cB * cN2 * 16) / 256, 256, 0, stream>>>(key,   (__bf16*)(ws + OFF_XKT));
  knn_kernel<<<dim3(cN / 256, cB), 256, 0, stream>>>(pos1, pos2, (int*)(ws + OFF_IDX));

  qkv_kernel<<<dim3(cN / 128, cB), 256, 0, stream>>>(
      (const __bf16*)(ws + OFF_XQT), (const __bf16*)(ws + OFF_WQF), bq,
      (float*)(ws + OFF_QT));
  qkv_kernel<<<dim3(cN2 / 128, cB), 256, 0, stream>>>(
      (const __bf16*)(ws + OFF_XKT), (const __bf16*)(ws + OFF_WKF), bk,
      (float*)(ws + OFF_KT));
  qkv_kernel<<<dim3(cN2 / 128, cB), 256, 0, stream>>>(
      (const __bf16*)(ws + OFF_XKT), (const __bf16*)(ws + OFF_WVF), bv,
      (float*)(ws + OFF_VT));

  attn_kernel<<<dim3(cN / 16, cB), 256, 0, stream>>>(
      pos1, pos2, query, pw1, pb2, abt, be, (const char*)ws, (float*)d_out);
}
